// Module_64029372448785
// MI455X (gfx1250) — matrix operand order repaired and checked by reading, not compiled, not run on hardware
//
#include <hip/hip_runtime.h>
#include <hip/hip_bf16.h>
#include <math.h>

#define DIM   256
#define NQ    512
#define NKK   512
#define TJ    16          // KWk rows per TDM tile

typedef _Float16     v16h __attribute__((ext_vector_type(16)));
typedef float        v8f  __attribute__((ext_vector_type(8)));
typedef unsigned int v4u  __attribute__((ext_vector_type(4)));
typedef int          v4i  __attribute__((ext_vector_type(4)));
typedef int          v8i  __attribute__((ext_vector_type(8)));

#ifndef __has_builtin
#define __has_builtin(x) 0
#endif
#if __has_builtin(__builtin_amdgcn_tensor_load_to_lds)
#define HAVE_TDM 1
#else
#define HAVE_TDM 0
#endif

__device__ __forceinline__ float waveRedSum(float v) {
#pragma unroll
    for (int off = 16; off > 0; off >>= 1) v += __shfl_xor(v, off, 32);
    return v;
}

// Low 32 bits of a flat pointer into the LDS aperture are the LDS byte offset
// (ISA 10.2).
__device__ __forceinline__ unsigned ldsAddr(const void* p) {
    return (unsigned)(unsigned long long)p;
}

#if HAVE_TDM
// Issue TDM load: `rows` x DIM f32 tile starting at gptr (row stride DIM) -> LDS.
// D# per ISA 8.3/8.4: group0 = {count=1, lds_addr, global_addr, type=2},
// group1 = {data_size=4B, tensor_dim0=DIM, tensor_dim1=NKK, tile_dim0=DIM,
//           tile_dim1=rows, tensor_dim0_stride=DIM}.
__device__ __forceinline__ void tdm_load_tile(unsigned lds_addr,
                                              const float* gptr, int rows)
{
    unsigned long long ga = (unsigned long long)gptr;
    v4u g0;
    g0[0] = 1u;                                           // count=1, user mode
    g0[1] = lds_addr;                                     // LDS byte address
    g0[2] = (unsigned)(ga & 0xffffffffu);                 // global_addr[31:0]
    g0[3] = (unsigned)((ga >> 32) & 0x01ffffffu) | 0x80000000u; // [56:32] | type=2

    v8i g1;
    g1[0] = (int)(2u << 16);                              // data_size=2 -> 4 bytes
    g1[1] = (int)((unsigned)DIM << 16);                   // tensor_dim0[15:0]
    g1[2] = (int)((unsigned)NKK << 16);                   // dim0[31:16]=0 | tensor_dim1[15:0]
    g1[3] = (int)((unsigned)DIM << 16);                   // dim1[31:16]=0 | tile_dim0
    g1[4] = rows;                                         // tile_dim1 | tile_dim2=0
    g1[5] = DIM;                                          // tensor_dim0_stride[31:0]
    g1[6] = 0;                                            // stride hi | dim1_stride lo
    g1[7] = 0;

    v4i z4a = {0, 0, 0, 0};
    v4i z4b = {0, 0, 0, 0};
#if __clang_major__ >= 23
    v8i z8 = {0, 0, 0, 0, 0, 0, 0, 0};
    __builtin_amdgcn_tensor_load_to_lds(g0, g1, z4a, z4b, z8, 0);
#else
    __builtin_amdgcn_tensor_load_to_lds(g0, g1, z4a, z4b, 0);
#endif
}
#endif

// ---------------------------------------------------------------------------
// C[m,n] = act( sum_c A[m,c] * B[n,c] + bias[n] )   (C = A @ B^T)
// One wave per 16x16 C tile via v_wmma_f32_16x16x32_f16 (f16 in, f32 acc).
// ---------------------------------------------------------------------------
__global__ void wmma_gemm_abt(const float* __restrict__ A, int lda,
                              const float* __restrict__ B, int ldb,
                              const float* __restrict__ bias,
                              float* __restrict__ C, int ldc,
                              int Kdim, int relu)
{
    const int lane  = threadIdx.x & 31;
    const int tn    = blockIdx.x * 16;
    const int tm    = blockIdx.y * 16;
    const int mrow  = tm + (lane & 15);
    const int ncol  = tn + (lane & 15);
    const int kselA = (lane & 16) ? 16 : 0;   // A frag: lanes16-31 hold K+16
    const int kselB = (lane & 16) ? 16 : 0;   // B frag: lanes16-31 hold K+16

    v8f acc = {0.f,0.f,0.f,0.f,0.f,0.f,0.f,0.f};

    for (int k0 = 0; k0 < Kdim; k0 += 32) {
        v16h a, b;
        const float* Ap = A + mrow * lda + k0;
#pragma unroll
        for (int h = 0; h < 16; h += 2) {
            const int k = kselA + h;   // contiguous along K
            float2 f = *(const float2*)(Ap + k);
            a[h]     = (_Float16)f.x;
            a[h + 1] = (_Float16)f.y;
        }
        const float* Bp = B + ncol * ldb + k0 + kselB;     // contiguous along K
#pragma unroll
        for (int h = 0; h < 16; h += 4) {
            float4 f = *(const float4*)(Bp + h);
            b[h]     = (_Float16)f.x;
            b[h + 1] = (_Float16)f.y;
            b[h + 2] = (_Float16)f.z;
            b[h + 3] = (_Float16)f.w;
        }
        acc = __builtin_amdgcn_wmma_f32_16x16x32_f16(false, a, false, b,
                                                     (short)0, acc, false, false);
    }
    const float bv = bias ? bias[ncol] : 0.0f;
#pragma unroll
    for (int r = 0; r < 8; ++r) {
        const int row = tm + r + ((lane & 16) ? 8 : 0);    // f32 C/D 16x16 layout
        float v = acc[r] + bv;
        if (relu) v = fmaxf(v, 0.0f);
        C[row * ldc + ncol] = v;
    }
}

// ---------------------------------------------------------------------------
// C = A @ B  (B row-major Kdim x N).  Used for context = weights @ V.
// ---------------------------------------------------------------------------
__global__ void wmma_gemm_ab(const float* __restrict__ A, int lda,
                             const float* __restrict__ B, int ldb,
                             float* __restrict__ C, int ldc, int Kdim)
{
    const int lane  = threadIdx.x & 31;
    const int tn    = blockIdx.x * 16;
    const int tm    = blockIdx.y * 16;
    const int mrow  = tm + (lane & 15);
    const int ncol  = tn + (lane & 15);
    const int kselA = (lane & 16) ? 16 : 0;
    const int kselB = (lane & 16) ? 16 : 0;

    v8f acc = {0.f,0.f,0.f,0.f,0.f,0.f,0.f,0.f};

    for (int k0 = 0; k0 < Kdim; k0 += 32) {
        v16h a, b;
        const float* Ap = A + mrow * lda + k0;
#pragma unroll
        for (int h = 0; h < 16; h += 2) {
            const int k = kselA + h;
            float2 f = *(const float2*)(Ap + k);
            a[h]     = (_Float16)f.x;
            a[h + 1] = (_Float16)f.y;
        }
        const float* Bp = B + (k0 + kselB) * ldb + ncol;   // B column, stride ldb
#pragma unroll
        for (int h = 0; h < 16; ++h)
            b[h] = (_Float16)Bp[h * ldb];
        acc = __builtin_amdgcn_wmma_f32_16x16x32_f16(false, a, false, b,
                                                     (short)0, acc, false, false);
    }
#pragma unroll
    for (int r = 0; r < 8; ++r) {
        const int row = tm + r + ((lane & 16) ? 8 : 0);
        C[row * ldc + ncol] = acc[r];
    }
}

// ---------------------------------------------------------------------------
// mu_prior[j] = h[j,:] . w2p + b2p - 0.5*sigma_prior^2   (sigma_prior = 0.5)
// ---------------------------------------------------------------------------
__global__ void muprior_kernel(const float* __restrict__ h,
                               const float* __restrict__ w2p,
                               const float* __restrict__ b2p,
                               float* __restrict__ mu_prior)
{
    __shared__ float sc[8];
    const int j = blockIdx.x, tid = threadIdx.x;
    float v = h[j * DIM + tid] * w2p[tid];
    v = waveRedSum(v);
    if ((tid & 31) == 0) sc[tid >> 5] = v;
    __syncthreads();
    if (tid == 0) {
        float t = 0.f;
        for (int w = 0; w < 8; ++w) t += sc[w];
        mu_prior[j] = t + b2p[0] - 0.125f;
    }
}

// ---------------------------------------------------------------------------
// One block per query row i (8 waves). KWk is streamed through LDS with the
// Tensor Data Mover (double-buffered, s_wait_tensorcnt), while per-lane
// QWq/w_mu/w_lv slices live in registers. Fused: pairwise ReLU-dot, KL
// partials, sampling, row softmax. Never materializes the 512x512x256 latent.
// ---------------------------------------------------------------------------
__global__ void pair_row_kernel(const float* __restrict__ QWq,
                                const float* __restrict__ KWk,
                                const float* __restrict__ w_mu,
                                const float* __restrict__ w_lv,
                                const float* __restrict__ b_mu,
                                const float* __restrict__ b_lv,
                                const float* __restrict__ mu_prior,
                                const float* __restrict__ eps,
                                float* __restrict__ Wt,
                                float* __restrict__ kl_rows)
{
    __shared__ float kbuf[2][TJ * DIM];   // 2 x 16KB double buffer
    __shared__ float srow[NKK];
    __shared__ float sc[8];
    const int i    = blockIdx.x;
    const int tid  = threadIdx.x;          // 256 threads = 8 waves
    const int lane = tid & 31;
    const int wave = tid >> 5;

    // registerize the lane's slices (reused for all 512 j-rows)
    float qv[8], wmu[8], wlv[8];
#pragma unroll
    for (int t = 0; t < 8; ++t) {
        const int d = lane + 32 * t;
        qv[t]  = QWq[i * DIM + d];
        wmu[t] = w_mu[d];
        wlv[t] = w_lv[d];
    }
    __builtin_prefetch(eps + i * NKK, 0, 0);   // global_prefetch_b8

    const float bmu = b_mu[0], blv = b_lv[0];
    float klacc = 0.f;

#if HAVE_TDM
    if (wave == 0) tdm_load_tile(ldsAddr(&kbuf[0][0]), KWk, TJ);
#endif
    int cur = 0;
    for (int t = 0; t < NKK / TJ; ++t) {
#if HAVE_TDM
        if (wave == 0) __builtin_amdgcn_s_wait_tensorcnt(0);
        __syncthreads();   // tile t ready; compute on other buffer finished
        if (wave == 0 && t + 1 < NKK / TJ)
            tdm_load_tile(ldsAddr(&kbuf[cur ^ 1][0]),
                          KWk + (t + 1) * TJ * DIM, TJ);
#else
        __syncthreads();
        for (int e = tid; e < TJ * DIM; e += 256)
            kbuf[cur][e] = KWk[t * TJ * DIM + e];
        __syncthreads();
#endif
#pragma unroll
        for (int jj = 0; jj < TJ / 8; ++jj) {
            const int jl = wave * (TJ / 8) + jj;
            const int j  = t * TJ + jl;
            const float* krow = &kbuf[cur][jl * DIM];
            float phi = 0.f, lv = 0.f;
#pragma unroll
            for (int d8 = 0; d8 < 8; ++d8) {
                const float r = fmaxf(qv[d8] + krow[lane + 32 * d8], 0.0f);
                phi = fmaf(r, wmu[d8], phi);
                lv  = fmaf(r, wlv[d8], lv);
            }
            phi = waveRedSum(phi) + bmu;
            lv  = waveRedSum(lv)  + blv;
            const float sig2 = __expf(lv);
            const float sig  = __expf(0.5f * lv);
            const float mu   = phi - 0.5f * sig2;
            const float dmu  = mu - mu_prior[j];
            // log(sigma_prior/sig) + (sig^2+dmu^2)/(2*sigma_prior^2) - 0.5
            const float kle = (-0.69314718f - 0.5f * lv)
                              + 2.0f * (sig2 + dmu * dmu) - 0.5f;
            const float s = __expf(fmaf(sig, eps[i * NKK + j], mu));  // TEMP=1
            if (lane == 0) { srow[j] = s; klacc += kle; }
        }
        cur ^= 1;
    }
    if (lane == 0) sc[wave] = klacc;
    __syncthreads();
    if (tid == 0) {
        float t = 0.f;
        for (int w = 0; w < 8; ++w) t += sc[w];
        kl_rows[i] = t;
    }
    __syncthreads();

    // ----- softmax over srow[0..511] -----
    float m = fmaxf(srow[tid], srow[tid + 256]);
#pragma unroll
    for (int off = 16; off > 0; off >>= 1) m = fmaxf(m, __shfl_xor(m, off, 32));
    if (lane == 0) sc[wave] = m;
    __syncthreads();
    if (tid == 0) {
        float t = sc[0];
        for (int w = 1; w < 8; ++w) t = fmaxf(t, sc[w]);
        sc[0] = t;
    }
    __syncthreads();
    m = sc[0];
    __syncthreads();

    const float e0 = __expf(srow[tid]       - m);
    const float e1 = __expf(srow[tid + 256] - m);
    float acc = waveRedSum(e0 + e1);
    if (lane == 0) sc[wave] = acc;
    __syncthreads();
    if (tid == 0) {
        float t = 0.f;
        for (int w = 0; w < 8; ++w) t += sc[w];
        sc[0] = t;
    }
    __syncthreads();
    const float inv = 1.0f / sc[0];
    Wt[i * NKK + tid]       = e0 * inv;
    Wt[i * NKK + tid + 256] = e1 * inv;
}

// ---------------------------------------------------------------------------
// Row LayerNorm of context (population variance, eps = 1e-5)
// ---------------------------------------------------------------------------
__global__ void ln_kernel(const float* __restrict__ ctx,
                          const float* __restrict__ gamma,
                          const float* __restrict__ beta,
                          float* __restrict__ out)
{
    __shared__ float sc[8], sc2[8];
    const int i = blockIdx.x, tid = threadIdx.x;
    const int lane = tid & 31, wave = tid >> 5;
    const float x  = ctx[i * DIM + tid];
    const float s  = waveRedSum(x);
    const float s2 = waveRedSum(x * x);
    if (lane == 0) { sc[wave] = s; sc2[wave] = s2; }
    __syncthreads();
    if (tid == 0) {
        float a = 0.f, b = 0.f;
        for (int w = 0; w < 8; ++w) { a += sc[w]; b += sc2[w]; }
        sc[0] = a; sc2[0] = b;
    }
    __syncthreads();
    const float mean = sc[0]  * (1.0f / DIM);
    const float var  = sc2[0] * (1.0f / DIM) - mean * mean;
    out[i * DIM + tid] = (x - mean) * rsqrtf(var + 1e-5f) * gamma[tid] + beta[tid];
}

__global__ void kl_finalize(const float* __restrict__ kl_rows,
                            float* __restrict__ out_kl)
{
    __shared__ float sc[8];
    const int tid = threadIdx.x;
    float v = kl_rows[tid] + kl_rows[tid + 256];
    v = waveRedSum(v);
    if ((tid & 31) == 0) sc[tid >> 5] = v;
    __syncthreads();
    if (tid == 0) {
        float t = 0.f;
        for (int w = 0; w < 8; ++w) t += sc[w];
        out_kl[0] = t / (float)(NQ * NKK);
    }
}

extern "C" void kernel_launch(void* const* d_in, const int* in_sizes, int n_in,
                              void* d_out, int out_size, void* d_ws, size_t ws_size,
                              hipStream_t stream) {
    const float* Q     = (const float*)d_in[0];
    const float* Km    = (const float*)d_in[1];
    const float* V     = (const float*)d_in[2];
    const float* eps   = (const float*)d_in[3];
    const float* W1p   = (const float*)d_in[4];
    const float* b1p   = (const float*)d_in[5];
    const float* w2p   = (const float*)d_in[6];
    const float* b2p   = (const float*)d_in[7];
    const float* Wpost = (const float*)d_in[8];
    const float* bpost = (const float*)d_in[9];
    const float* w_mu  = (const float*)d_in[10];
    const float* b_mu  = (const float*)d_in[11];
    const float* w_lv  = (const float*)d_in[12];
    const float* b_lv  = (const float*)d_in[13];
    const float* gamma = (const float*)d_in[14];
    const float* beta  = (const float*)d_in[15];

    float* ws       = (float*)d_ws;
    float* QWq      = ws;                    // 512*256
    float* KWk      = ws + 131072;           // 512*256
    float* h        = ws + 262144;           // 512*256
    float* mu_prior = ws + 393216;           // 512
    float* Wt       = ws + 393728;           // 512*512
    float* ctx      = ws + 655872;           // 512*256
    float* kl_rows  = ws + 786944;           // 512
    float* out      = (float*)d_out;         // 512*256 + 1 (kl)

    const dim3 b32(32);
    const dim3 gemmGridND(DIM / 16, NQ / 16);   // N=256 x M=512 tiles

    // QWq = Q @ Wq^T            (Wq = Wpost[:, :256], ld = 512)
    wmma_gemm_abt<<<gemmGridND, b32, 0, stream>>>(Q, DIM, Wpost, 2 * DIM,
                                                  nullptr, QWq, DIM, DIM, 0);
    // KWk = K @ Wk^T + bpost    (Wk = Wpost[:, 256:], ld = 512)
    wmma_gemm_abt<<<gemmGridND, b32, 0, stream>>>(Km, DIM, Wpost + DIM, 2 * DIM,
                                                  bpost, KWk, DIM, DIM, 0);
    // h = relu(K @ W1p^T + b1p)
    wmma_gemm_abt<<<gemmGridND, b32, 0, stream>>>(Km, DIM, W1p, DIM,
                                                  b1p, h, DIM, DIM, 1);
    // mu_prior = h @ w2p + b2p - 0.125
    muprior_kernel<<<NKK, 256, 0, stream>>>(h, w2p, b2p, mu_prior);

    // fused pairwise latent reduction + KL + sampling + softmax (TDM-fed)
    pair_row_kernel<<<NQ, 256, 0, stream>>>(QWq, KWk, w_mu, w_lv, b_mu, b_lv,
                                            mu_prior, eps, Wt, kl_rows);

    // context = weights @ V
    wmma_gemm_ab<<<dim3(DIM / 16, NQ / 16), b32, 0, stream>>>(Wt, NKK, V, DIM,
                                                              ctx, DIM, NKK);
    // layernorm -> out
    ln_kernel<<<NQ, 256, 0, stream>>>(ctx, gamma, beta, out);
    // kl scalar -> out[512*256]
    kl_finalize<<<1, 256, 0, stream>>>(kl_rows, out + NQ * DIM);
}
